// AttentionBlock_32298154066358
// MI455X (gfx1250) — matrix operand order repaired and checked by reading, not compiled, not run on hardware
//
#include <hip/hip_runtime.h>
#include <hip/hip_bf16.h>

typedef __bf16 bf16_t;
typedef __attribute__((ext_vector_type(16))) __bf16 v16bf;
typedef __attribute__((ext_vector_type(8)))  __bf16 v8bf;
typedef __attribute__((ext_vector_type(8)))  float  v8f;
typedef __attribute__((ext_vector_type(4)))  float  v4f;

namespace {
constexpr int   kB   = 2;
constexpr int   kC   = 256;
constexpr int   kT   = 4096;   // 64*64
constexpr int   kNH  = 4;
constexpr int   kCH  = 64;     // C / NH
constexpr int   kG   = 32;
constexpr int   kCPG = 8;      // C / G
constexpr float kEps = 1e-5f;
constexpr float kQKScale = 0.35355339059327373f; // 64^-0.25 = 1/sqrt(sqrt(ch))
}

__device__ __forceinline__ v8f wmma_bf16(v16bf a, v16bf b, v8f c) {
  // D = A(16x32 bf16) * B(32x16 bf16) + C(16x16 f32)
  return __builtin_amdgcn_wmma_f32_16x16x32_bf16(false, a, false, b, (short)0, c,
                                                 false, false);
}

__device__ __forceinline__ v16bf pack16(v8bf lo, v8bf hi) {
  v16bf r;
#pragma unroll
  for (int i = 0; i < 8; ++i) { r[i] = lo[i]; r[i + 8] = hi[i]; }
  return r;
}

// A-fragment (16x32 bf16) from a row-major bf16 row: element e covers
// K = e + hi*16 -> one 32B run.
__device__ __forceinline__ v16bf load_afrag(const bf16_t* row, int hi) {
  return pack16(*(const v8bf*)(row + hi * 16), *(const v8bf*)(row + hi * 16 + 8));
}

// B-fragment (32x16 bf16) from a row-major bf16 row (the "row" is this lane's
// column of the B matrix): element e covers K = e + hi*16 -> one 32B run.
__device__ __forceinline__ v16bf load_bfrag(const bf16_t* row, int hi) {
  return pack16(*(const v8bf*)(row + hi * 16), *(const v8bf*)(row + hi * 16 + 8));
}

// ---------------------------------------------------------------------------
// Kernel 0: one-shot f32 -> bf16 weight conversion (vectorized 4 at a time).
// ---------------------------------------------------------------------------
__global__ __launch_bounds__(256) void cvt_w_kernel(const float* __restrict__ src,
                                                    bf16_t* __restrict__ dst, int n4) {
  const int i = blockIdx.x * 256 + threadIdx.x;
  if (i < n4) {
    const v4f v = *(const v4f*)(src + i * 4);
    bf16_t* d = dst + i * 4;
#pragma unroll
    for (int k = 0; k < 4; ++k) d[k] = (bf16_t)v[k];
  }
}

// ---------------------------------------------------------------------------
// Kernel 1: GroupNorm statistics. One block per (b, g); group channels are
// contiguous in x[b][c][t], so the group is one flat run of 8*4096 floats.
// ---------------------------------------------------------------------------
__global__ __launch_bounds__(256) void gn_stats_kernel(const float* __restrict__ x,
                                                       float* __restrict__ stats) {
  const int bg = blockIdx.x;                       // b*32 + g
  const float* base = x + (size_t)bg * kCPG * kT;  // (b*C + g*8)*T
  float s = 0.f, ss = 0.f;
  for (int i = threadIdx.x; i < kCPG * kT; i += 256) {
    const float v = base[i];
    s += v; ss += v * v;
  }
  __shared__ float shs[256];
  __shared__ float shq[256];
  shs[threadIdx.x] = s; shq[threadIdx.x] = ss;
  __syncthreads();
  for (int off = 128; off > 0; off >>= 1) {
    if (threadIdx.x < off) {
      shs[threadIdx.x] += shs[threadIdx.x + off];
      shq[threadIdx.x] += shq[threadIdx.x + off];
    }
    __syncthreads();
  }
  if (threadIdx.x == 0) {
    const float invN = 1.0f / (float)(kCPG * kT);
    const float mean = shs[0] * invN;
    const float var  = shq[0] * invN - mean * mean;
    stats[bg * 2]     = mean;
    stats[bg * 2 + 1] = rsqrtf(var + kEps);
  }
}

// ---------------------------------------------------------------------------
// Kernel 2: apply GroupNorm affine, transpose [c][t] -> [t][c], store bf16.
// xnT layout makes every later GEMM B-fragment a contiguous 16B load.
// ---------------------------------------------------------------------------
__global__ __launch_bounds__(256) void gn_apply_kernel(const float* __restrict__ x,
                                                       const float* __restrict__ stats,
                                                       const float* __restrict__ gamma,
                                                       const float* __restrict__ beta,
                                                       bf16_t* __restrict__ xnT) {
  __shared__ float tile[32][33];
  const int b  = blockIdx.z;
  const int c0 = blockIdx.y * 32;
  const int t0 = blockIdx.x * 32;
  const int tx = threadIdx.x;   // 0..31
  const int ty = threadIdx.y;   // 0..7
#pragma unroll
  for (int k = 0; k < 4; ++k) {
    const int c = c0 + ty + k * 8;
    const int g = c >> 3;
    const float mu = stats[(b * kG + g) * 2];
    const float rs = stats[(b * kG + g) * 2 + 1];
    const float v  = x[((size_t)b * kC + c) * kT + t0 + tx];
    tile[ty + k * 8][tx] = (v - mu) * rs * gamma[c] + beta[c];
  }
  __syncthreads();
#pragma unroll
  for (int k = 0; k < 4; ++k) {
    const int tl = ty + k * 8;
    xnT[((size_t)b * kT + t0 + tl) * kC + c0 + tx] = (bf16_t)tile[tx][tl];
  }
}

// ---------------------------------------------------------------------------
// Kernel 3: QKV 1x1 conv as WMMA GEMM: qkv[o,t] = sum_c W[o,c] * xnT[t,c].
// One wave per 16x16 output tile; K loop of 8 x (K=32) bf16 WMMAs with all
// fragments as contiguous b128 loads (weights pre-converted to bf16).
// Epilogue scatters rows into QT/KT (scaled) and V per head.
// ---------------------------------------------------------------------------
__global__ __launch_bounds__(128) void qkv_kernel(const bf16_t* __restrict__ Wq,
                                                  const float* __restrict__ bq,
                                                  const bf16_t* __restrict__ xnT,
                                                  bf16_t* __restrict__ QT,
                                                  bf16_t* __restrict__ KT,
                                                  bf16_t* __restrict__ Vm) {
  const int lane = threadIdx.x & 31;
  const int wave = threadIdx.x >> 5;
  const int ln   = lane & 15;
  const int hi   = lane >> 4;
  const int tile = blockIdx.x * 4 + wave;
  const int b    = tile / (48 * 256);
  const int rem  = tile % (48 * 256);
  const int mt   = rem / 256;          // 0..47  (o tile, 3C/16)
  const int nt   = rem % 256;          // 0..255 (t tile)
  const int t    = nt * 16 + ln;

  const bf16_t* wr = Wq + (size_t)(mt * 16 + ln) * kC;     // A: row o = mt*16+ln
  const bf16_t* xr = xnT + ((size_t)b * kT + t) * kC;      // B: col t

  v8f acc = {};
#pragma unroll
  for (int kb = 0; kb < 8; ++kb) {
    const int cb = kb * 32;
    const v16bf a  = load_afrag(wr + cb, hi);
    const v16bf xb = load_bfrag(xr + cb, hi);
    acc = wmma_bf16(a, xb, acc);
  }

#pragma unroll
  for (int r = 0; r < 8; ++r) {
    const int o = mt * 16 + r + hi * 8;       // C/D layout: M = r + hi*8
    const float v = acc[r] + bq[o];
    const int h  = o / (3 * kCH);
    const int j  = o % (3 * kCH);
    const int bh = b * kNH + h;
    if (j < kCH) {
      QT[((size_t)bh * kT + t) * kCH + j] = (bf16_t)(v * kQKScale);
    } else if (j < 2 * kCH) {
      KT[((size_t)bh * kT + t) * kCH + (j - kCH)] = (bf16_t)(v * kQKScale);
    } else {
      Vm[((size_t)bh * kCH + (j - 2 * kCH)) * kT + t] = (bf16_t)v;
    }
  }
}

// ---------------------------------------------------------------------------
// Kernel 4: flash attention. One wave owns 16 queries, streams keys in blocks
// of 32. Computes S^T = K^T*Q (softmax rows live inside a lane: 16 regs + one
// shfl_xor(16)), builds the P B-fragment with one cross-half shuffle, then
// O^T += V * P via 4 WMMAs. No score matrix ever touches memory.
// ---------------------------------------------------------------------------
__global__ __launch_bounds__(128) void attn_kernel(const bf16_t* __restrict__ QT,
                                                   const bf16_t* __restrict__ KT,
                                                   const bf16_t* __restrict__ Vm,
                                                   bf16_t* __restrict__ OT) {
  const int lane = threadIdx.x & 31;
  const int wave = threadIdx.x >> 5;
  const int ln   = lane & 15;
  const int hi   = lane >> 4;
  const int bh   = blockIdx.y;        // 0..7
  const int b    = bh >> 2;
  const int h    = bh & 3;
  const int t    = blockIdx.x * 64 + wave * 16 + ln;   // global query index

  const bf16_t* QTbh = QT + (size_t)bh * kT * kCH;
  const bf16_t* KTbh = KT + (size_t)bh * kT * kCH;
  const bf16_t* Vbh  = Vm + (size_t)bh * kCH * kT;

  // Q as two B fragments (c = 0..31 and 32..63), column t, contiguous loads.
  v16bf qb0, qb1;
  {
    const bf16_t* qr = QTbh + (size_t)t * kCH;
    qb0 = load_bfrag(qr, hi);
    qb1 = load_bfrag(qr + 32, hi);
  }

  v8f oacc[4] = {{}, {}, {}, {}};     // O^T tiles, c = ct*16 .. ct*16+15
  float mrow = -__builtin_inff();
  float lrow = 0.f;

  for (int sb = 0; sb < kT; sb += 32) {
    // S^T fragments: s[f] holds S[s = sb + f*16 + r + hi*8][t = ln]
    v8f s[2];
#pragma unroll
    for (int f = 0; f < 2; ++f) {
      const bf16_t* kr = KTbh + (size_t)(sb + f * 16 + ln) * kCH;  // A row = key s
      const v16bf ka0 = load_afrag(kr, hi);
      const v16bf ka1 = load_afrag(kr + 32, hi);
      v8f acc = {};
      acc = wmma_bf16(ka0, qb0, acc);
      acc = wmma_bf16(ka1, qb1, acc);
      s[f] = acc;
    }

    // Online softmax (per query t; both half-waves hold the same t).
    float mx = s[0][0];
#pragma unroll
    for (int r = 1; r < 8; ++r) mx = fmaxf(mx, s[0][r]);
#pragma unroll
    for (int r = 0; r < 8; ++r) mx = fmaxf(mx, s[1][r]);
    mx = fmaxf(mx, __shfl_xor(mx, 16, 32));
    const float mnew  = fmaxf(mrow, mx);
    const float alpha = __expf(mrow - mnew);

    float p0[8], p1[8];
    float rsum = 0.f;
#pragma unroll
    for (int r = 0; r < 8; ++r) {
      p0[r] = __expf(s[0][r] - mnew);
      p1[r] = __expf(s[1][r] - mnew);
      rsum += p0[r] + p1[r];
    }
    rsum += __shfl_xor(rsum, 16, 32);
    lrow = lrow * alpha + rsum;
    mrow = mnew;

    // P as B fragment (k = s, n = t): element e needs s = e + hi*16.
    // Own regs cover s = {hi*8..hi*8+7, 16+hi*8..}; fetch the rest from the
    // partner half-wave with one shfl_xor(16) per register.
    v16bf pb;
#pragma unroll
    for (int r = 0; r < 8; ++r) {
      const float q0 = __shfl_xor(p0[r], 16, 32);   // partner's s = r + 8*(1-hi)
      const float q1 = __shfl_xor(p1[r], 16, 32);
      pb[r]     = (bf16_t)(hi ? q1 : p0[r]);
      pb[r + 8] = (bf16_t)(hi ? p1[r] : q0);
    }

    // Rescale running output, then O^T += V * P
#pragma unroll
    for (int ct = 0; ct < 4; ++ct) {
#pragma unroll
      for (int r = 0; r < 8; ++r) oacc[ct][r] *= alpha;
    }
#pragma unroll
    for (int ct = 0; ct < 4; ++ct) {
      const bf16_t* vr = Vbh + (size_t)(ct * 16 + ln) * kT + sb;   // A row = channel
      const v16bf va = load_afrag(vr, hi);
      oacc[ct] = wmma_bf16(va, pb, oacc[ct]);
    }
  }

  // Normalize and store O^T as OT[b][t][C] bf16 (contiguous 16B stores).
  const float inv = 1.0f / lrow;
  bf16_t* orow = OT + ((size_t)b * kT + t) * kC + h * kCH;
#pragma unroll
  for (int ct = 0; ct < 4; ++ct) {
    v8bf w;
#pragma unroll
    for (int r = 0; r < 8; ++r) w[r] = (bf16_t)(oacc[ct][r] * inv);
    *(v8bf*)(orow + ct * 16 + hi * 8) = w;
  }
}

// ---------------------------------------------------------------------------
// Kernel 5: proj 1x1 conv (WMMA GEMM) + bias + residual, f32 output.
// ---------------------------------------------------------------------------
__global__ __launch_bounds__(128) void proj_kernel(const bf16_t* __restrict__ Wp,
                                                   const float* __restrict__ bp,
                                                   const bf16_t* __restrict__ OT,
                                                   const float* __restrict__ x,
                                                   float* __restrict__ out) {
  const int lane = threadIdx.x & 31;
  const int wave = threadIdx.x >> 5;
  const int ln   = lane & 15;
  const int hi   = lane >> 4;
  const int tile = blockIdx.x * 4 + wave;
  const int b    = tile / (16 * 256);
  const int rem  = tile % (16 * 256);
  const int mt   = rem / 256;          // 0..15 (o tile)
  const int nt   = rem % 256;          // 0..255 (t tile)
  const int t    = nt * 16 + ln;

  const bf16_t* wr = Wp + (size_t)(mt * 16 + ln) * kC;
  const bf16_t* ar = OT + ((size_t)b * kT + t) * kC;

  v8f acc = {};
#pragma unroll
  for (int kb = 0; kb < 8; ++kb) {
    const int cb = kb * 32;
    const v16bf a  = load_afrag(wr + cb, hi);
    const v16bf bb = load_bfrag(ar + cb, hi);
    acc = wmma_bf16(a, bb, acc);
  }

#pragma unroll
  for (int r = 0; r < 8; ++r) {
    const int o = mt * 16 + r + hi * 8;
    const size_t idx = ((size_t)b * kC + o) * kT + t;
    out[idx] = x[idx] + acc[r] + bp[o];
  }
}

// ---------------------------------------------------------------------------
extern "C" void kernel_launch(void* const* d_in, const int* in_sizes, int n_in,
                              void* d_out, int out_size, void* d_ws, size_t ws_size,
                              hipStream_t stream) {
  (void)in_sizes; (void)n_in; (void)out_size; (void)ws_size;
  const float* x        = (const float*)d_in[0];
  const float* gn_scale = (const float*)d_in[1];
  const float* gn_bias  = (const float*)d_in[2];
  const float* w_qkv    = (const float*)d_in[3];
  const float* b_qkv    = (const float*)d_in[4];
  const float* w_proj   = (const float*)d_in[5];
  const float* b_proj   = (const float*)d_in[6];
  float* out = (float*)d_out;

  // Workspace carve-up (all 16B aligned): ~21 MB total.
  char* ws = (char*)d_ws;
  float* stats = (float*)ws;                                     // B*G*2 f32
  size_t off = 1024;
  bf16_t* xnT = (bf16_t*)(ws + off); off += (size_t)kB * kT * kC * 2;          // 4MB
  bf16_t* QT  = (bf16_t*)(ws + off); off += (size_t)kB * kNH * kT * kCH * 2;   // 4MB
  bf16_t* KT  = (bf16_t*)(ws + off); off += (size_t)kB * kNH * kT * kCH * 2;   // 4MB
  bf16_t* Vm  = (bf16_t*)(ws + off); off += (size_t)kB * kNH * kCH * kT * 2;   // 4MB
  bf16_t* OT  = (bf16_t*)(ws + off); off += (size_t)kB * kT * kC * 2;          // 4MB
  bf16_t* Wqb = (bf16_t*)(ws + off); off += (size_t)3 * kC * kC * 2;           // 384KB
  bf16_t* Wpb = (bf16_t*)(ws + off);                                           // 128KB

  const int nq4 = (3 * kC * kC) / 4;
  const int np4 = (kC * kC) / 4;
  cvt_w_kernel<<<(nq4 + 255) / 256, 256, 0, stream>>>(w_qkv, Wqb, nq4);
  cvt_w_kernel<<<(np4 + 255) / 256, 256, 0, stream>>>(w_proj, Wpb, np4);
  gn_stats_kernel<<<kB * kG, 256, 0, stream>>>(x, stats);
  gn_apply_kernel<<<dim3(kT / 32, kC / 32, kB), dim3(32, 8), 0, stream>>>(
      x, stats, gn_scale, gn_bias, xnT);
  qkv_kernel<<<(kB * 48 * 256) / 4, 128, 0, stream>>>(Wqb, b_qkv, xnT, QT, KT, Vm);
  attn_kernel<<<dim3(kT / 64, kB * kNH), 128, 0, stream>>>(QT, KT, Vm, OT);
  proj_kernel<<<(kB * 16 * 256) / 4, 128, 0, stream>>>(w_proj 
      ? Wpb : Wpb, b_proj, OT, x, out);
}
